// MLP_77111842832415
// MI455X (gfx1250) — matrix operand order repaired and checked by reading, not compiled, not run on hardware
//
#include <hip/hip_runtime.h>
#include <stdint.h>

// ---------------------------------------------------------------------------
// SNN block: [conv1x1 -> BN -> LIF] x2 on MI455X (gfx1250, wave32, WMMA).
// GEMMs in bf16 with f32 accumulate via v_wmma_f32_16x16x32_bf16.
// BN + LIF (T=4) fused into the GEMM epilogue; spikes stored as bf16 {0,1}.
// All WMMA operand traffic forced to addrspace(1) -> global_load_b128
// (avoids flat_load's dual LOADcnt/DScnt accounting).
// Workspace layout (bytes):
//   [0)            A1  : x packed bf16, pixel-major [T*B*196][384]   19,267,584
//   [19267584)     H   : spikes bf16,   pixel-major [T*B*196][1536]  77,070,336
//   [96337920)     W1b : w1 bf16 [1536][384]                          1,179,648
//   [97517568)     W2b : w2 bf16 [384][1536]                          1,179,648
//   total 98,697,216 bytes (~94.1 MB)
// ---------------------------------------------------------------------------

typedef __bf16 bf8v  __attribute__((ext_vector_type(8)));
typedef __bf16 bf16v __attribute__((ext_vector_type(16)));
typedef float  f8v   __attribute__((ext_vector_type(8)));
typedef unsigned int u32x4 __attribute__((ext_vector_type(4)));
typedef unsigned int u32x2 __attribute__((ext_vector_type(2)));

#define GAS __attribute__((address_space(1)))

#define T_  4
#define B_  32
#define C_  384
#define CH_ 1536
#define HW_ 196
#define HWT 13          // ceil(196/16) pixel tiles per image
#define EPS_ 1e-5f

// Load two 16B chunks as one 16x(2x8) WMMA half-operand; force global addrspace.
static __device__ __forceinline__ bf16v ld_ab(const __bf16* p0, const __bf16* p1) {
  const GAS bf8v* q0 = (const GAS bf8v*)(unsigned long long)p0;
  const GAS bf8v* q1 = (const GAS bf8v*)(unsigned long long)p1;
  bf8v a = *q0;
  bf8v b = *q1;
  bf16v r;
#pragma unroll
  for (int i = 0; i < 8; ++i) { r[i] = a[i]; r[i + 8] = b[i]; }
  return r;
}

static __device__ __forceinline__ void st_gbf(__bf16* p, __bf16 v) {
  *(GAS __bf16*)(unsigned long long)p = v;
}
static __device__ __forceinline__ void st_gf32(float* p, float v) {
  *(GAS float*)(unsigned long long)p = v;
}

static __device__ __forceinline__ f8v wmma_bf16(bf16v a, bf16v b, f8v c) {
  // D = A(16x32 bf16) * B(32x16 bf16) + C(16x16 f32)
  return __builtin_amdgcn_wmma_f32_16x16x32_bf16(false, a, false, b, (short)0, c,
                                                 false, false);
}

// ---------------- pack x: [T,B,C,H,W] f32 -> pixel-major bf16 [TB*196][384] --
__global__ __launch_bounds__(224) void snn_pack_x(const float* __restrict__ x,
                                                  __bf16* __restrict__ A1) {
  const int tb = blockIdx.x;          // 0..127
  const int hw = threadIdx.x;         // 0..223
  if (hw >= HW_) return;
  const float* src = x + ((size_t)tb * C_) * HW_ + hw;   // stride HW_ per channel
  __bf16* dst = A1 + ((size_t)tb * HW_ + hw) * C_;
#pragma unroll 2
  for (int c0 = 0; c0 < C_; c0 += 8) {
    union { __bf16 h[8]; u32x4 v; } u;
#pragma unroll
    for (int j = 0; j < 8; ++j)
      u.h[j] = (__bf16)src[(size_t)(c0 + j) * HW_];      // coalesced across lanes
    *(GAS u32x4*)(unsigned long long)(dst + c0) = u.v;   // 16B global store
  }
}

// ---------------- pack weights: f32 -> bf16 ---------------------------------
__global__ __launch_bounds__(256) void snn_pack_w(const float* __restrict__ w,
                                                  __bf16* __restrict__ wb, int n4) {
  int i = blockIdx.x * 256 + threadIdx.x;
  if (i >= n4) return;
  const GAS float* src = (const GAS float*)(unsigned long long)(w + (size_t)i * 4);
  union { __bf16 h[4]; u32x2 v; } u;
#pragma unroll
  for (int j = 0; j < 4; ++j) u.h[j] = (__bf16)src[j];
  *(GAS u32x2*)(unsigned long long)(wb + (size_t)i * 4) = u.v;
}

// ---------------- layer 1: GEMM(x,w1) + BN1 + LIF -> spikes H ---------------
// grid.x = B_*HWT (b, hw-tile), grid.y = 6 (256 out-ch per block), 8 waves.
// Wave tile: M=16 pixels x N=32 channels, 4 timesteps fused -> 8 WMMA / 12 loads.
__global__ __launch_bounds__(256) void snn_gemm1_lif(
    const __bf16* __restrict__ A1, const __bf16* __restrict__ W1b,
    const float* __restrict__ b1, const float* __restrict__ g1,
    const float* __restrict__ be1, const float* __restrict__ m1,
    const float* __restrict__ v1, __bf16* __restrict__ H) {
  const int lane = threadIdx.x & 31;
  const int wave = threadIdx.x >> 5;
  const int n    = lane & 15;      // tile column
  const int half = lane >> 4;
  const int b    = blockIdx.x / HWT;
  const int hw0  = (blockIdx.x % HWT) * 16;
  const int o0   = (blockIdx.y * 8 + wave) * 32;

  // A operand: lane row M = n ; K-runs at half*8 and half*8+16 (ISA layout)
  int hwA = hw0 + n; if (hwA > HW_ - 1) hwA = HW_ - 1;   // clamp padded rows
  const __bf16* ap0 = A1 + ((size_t)((0 * B_ + b) * HW_ + hwA)) * C_ + half * 8;
  const __bf16* ap1 = A1 + ((size_t)((1 * B_ + b) * HW_ + hwA)) * C_ + half * 8;
  const __bf16* ap2 = A1 + ((size_t)((2 * B_ + b) * HW_ + hwA)) * C_ + half * 8;
  const __bf16* ap3 = A1 + ((size_t)((3 * B_ + b) * HW_ + hwA)) * C_ + half * 8;
  // B operand: lane column ; K-runs at half*8 and half*8+16 ; two N-tiles
  const __bf16* bpA = W1b + (size_t)(o0 + n) * C_ + half * 8;
  const __bf16* bpB = W1b + (size_t)(o0 + 16 + n) * C_ + half * 8;

  f8v acc[T_][2];
#pragma unroll
  for (int t = 0; t < T_; ++t)
#pragma unroll
    for (int u = 0; u < 2; ++u)
#pragma unroll
      for (int i = 0; i < 8; ++i) acc[t][u][i] = 0.0f;

#pragma unroll 3
  for (int k = 0; k < C_; k += 32) {
    __builtin_prefetch(bpA + 64, 0, 1);                  // global_prefetch_b8
    bf16v bmA = ld_ab(bpA, bpA + 16);  bpA += 32;
    bf16v bmB = ld_ab(bpB, bpB + 16);  bpB += 32;
    bf16v a0 = ld_ab(ap0, ap0 + 16);  ap0 += 32;
    acc[0][0] = wmma_bf16(a0, bmA, acc[0][0]);
    acc[0][1] = wmma_bf16(a0, bmB, acc[0][1]);
    bf16v a1 = ld_ab(ap1, ap1 + 16);  ap1 += 32;
    acc[1][0] = wmma_bf16(a1, bmA, acc[1][0]);
    acc[1][1] = wmma_bf16(a1, bmB, acc[1][1]);
    bf16v a2 = ld_ab(ap2, ap2 + 16);  ap2 += 32;
    acc[2][0] = wmma_bf16(a2, bmA, acc[2][0]);
    acc[2][1] = wmma_bf16(a2, bmB, acc[2][1]);
    bf16v a3 = ld_ab(ap3, ap3 + 16);  ap3 += 32;
    acc[3][0] = wmma_bf16(a3, bmA, acc[3][0]);
    acc[3][1] = wmma_bf16(a3, bmB, acc[3][1]);
  }

  // BN + LIF across T, store spikes. D layout: VGPR r -> pixel r + 8*half.
#pragma unroll
  for (int u = 0; u < 2; ++u) {
    const int o = o0 + u * 16 + n;   // this lane's output channel (N = lane%16)
    const float sc = g1[o] * rsqrtf(v1[o] + EPS_);
    const float sh = be1[o] - m1[o] * sc + b1[o] * sc;   // conv bias folded in
#pragma unroll
    for (int r = 0; r < 8; ++r) {
      const int hw = hw0 + r + half * 8;
      const bool ok = hw < HW_;
      float vm = 0.0f;
#pragma unroll
      for (int t = 0; t < T_; ++t) {
        const float y = acc[t][u][r] * sc + sh;
        vm = 0.5f * (vm + y);                   // v += (y - v)/tau, tau=2
        const float s = (vm >= 1.0f) ? 1.0f : 0.0f;
        vm = (s != 0.0f) ? 0.0f : vm;           // hard reset to 0
        if (ok)
          st_gbf(&H[((size_t)(t * B_ + b) * HW_ + hw) * CH_ + o], (__bf16)s);
      }
    }
  }
}

// ---------------- layer 2: GEMM(H,w2) + BN2 + LIF -> out (f32) --------------
// Roles swapped: M=channels, N=pixels -> coalesced f32 output stores.
// grid.x = B_*HWT (b, hw-tile), grid.y = 2 (192 out-ch per block), 6 waves.
// Wave tile: M=32 channels x N=16 pixels, 4 timesteps fused -> 8 WMMA / 12 loads.
__global__ __launch_bounds__(192) void snn_gemm2_lif(
    const __bf16* __restrict__ H, const __bf16* __restrict__ W2b,
    const float* __restrict__ b2, const float* __restrict__ g2,
    const float* __restrict__ be2, const float* __restrict__ m2,
    const float* __restrict__ v2, float* __restrict__ out) {
  const int lane = threadIdx.x & 31;
  const int wave = threadIdx.x >> 5;
  const int n    = lane & 15;
  const int half = lane >> 4;
  const int b    = blockIdx.x / HWT;
  const int hw0  = (blockIdx.x % HWT) * 16;
  const int o0   = (blockIdx.y * 6 + wave) * 32;

  // A operand = w2 rows: lane rows o0+n / o0+16+n ; K-runs at half*8, half*8+16
  const __bf16* apA = W2b + (size_t)(o0 + n) * CH_ + half * 8;
  const __bf16* apB = W2b + (size_t)(o0 + 16 + n) * CH_ + half * 8;
  // B operand = spikes: lane column = pixel hw0+n ; K-runs at half*8, half*8+16
  int hwB = hw0 + n; if (hwB > HW_ - 1) hwB = HW_ - 1;
  const __bf16* bp0 = H + ((size_t)(0 * B_ + b) * HW_ + hwB) * CH_ + half * 8;
  const __bf16* bp1 = H + ((size_t)(1 * B_ + b) * HW_ + hwB) * CH_ + half * 8;
  const __bf16* bp2 = H + ((size_t)(2 * B_ + b) * HW_ + hwB) * CH_ + half * 8;
  const __bf16* bp3 = H + ((size_t)(3 * B_ + b) * HW_ + hwB) * CH_ + half * 8;

  f8v acc[T_][2];
#pragma unroll
  for (int t = 0; t < T_; ++t)
#pragma unroll
    for (int u = 0; u < 2; ++u)
#pragma unroll
      for (int i = 0; i < 8; ++i) acc[t][u][i] = 0.0f;

#pragma unroll 2
  for (int k = 0; k < CH_; k += 32) {
    __builtin_prefetch(apA + 64, 0, 1);                  // global_prefetch_b8
    bf16v amA = ld_ab(apA, apA + 16);  apA += 32;
    bf16v amB = ld_ab(apB, apB + 16);  apB += 32;
    bf16v b0 = ld_ab(bp0, bp0 + 16);    bp0 += 32;
    acc[0][0] = wmma_bf16(amA, b0, acc[0][0]);
    acc[0][1] = wmma_bf16(amB, b0, acc[0][1]);
    bf16v b1v = ld_ab(bp1, bp1 + 16);   bp1 += 32;
    acc[1][0] = wmma_bf16(amA, b1v, acc[1][0]);
    acc[1][1] = wmma_bf16(amB, b1v, acc[1][1]);
    bf16v b2v = ld_ab(bp2, bp2 + 16);   bp2 += 32;
    acc[2][0] = wmma_bf16(amA, b2v, acc[2][0]);
    acc[2][1] = wmma_bf16(amB, b2v, acc[2][1]);
    bf16v b3v = ld_ab(bp3, bp3 + 16);   bp3 += 32;
    acc[3][0] = wmma_bf16(amA, b3v, acc[3][0]);
    acc[3][1] = wmma_bf16(amB, b3v, acc[3][1]);
  }

  // Epilogue: lane column = pixel hw0+n ; VGPR r -> channel o0 + u*16 + r + 8*half.
  const int hw = hw0 + n;
  const bool ok = hw < HW_;
#pragma unroll
  for (int u = 0; u < 2; ++u) {
#pragma unroll
    for (int r = 0; r < 8; ++r) {
      const int o = o0 + u * 16 + r + 8 * half;
      const float sc = g2[o] * rsqrtf(v2[o] + EPS_);
      const float sh = be2[o] - m2[o] * sc + b2[o] * sc;
      float vm = 0.0f;
#pragma unroll
      for (int t = 0; t < T_; ++t) {
        const float y = acc[t][u][r] * sc + sh;
        vm = 0.5f * (vm + y);
        const float s = (vm >= 1.0f) ? 1.0f : 0.0f;
        vm = (s != 0.0f) ? 0.0f : vm;
        if (ok)  // lanes 0-15 write 64B-contiguous runs
          st_gf32(&out[(((size_t)(t * B_ + b)) * C_ + o) * HW_ + hw], s);
      }
    }
  }
}

// ---------------------------------------------------------------------------
extern "C" void kernel_launch(void* const* d_in, const int* in_sizes, int n_in,
                              void* d_out, int out_size, void* d_ws, size_t ws_size,
                              hipStream_t stream) {
  (void)in_sizes; (void)n_in; (void)out_size;
  const float* x  = (const float*)d_in[0];
  const float* w1 = (const float*)d_in[1];
  const float* b1 = (const float*)d_in[2];
  const float* g1 = (const float*)d_in[3];
  const float* be1= (const float*)d_in[4];
  const float* m1 = (const float*)d_in[5];
  const float* v1 = (const float*)d_in[6];
  const float* w2 = (const float*)d_in[7];
  const float* b2 = (const float*)d_in[8];
  const float* g2 = (const float*)d_in[9];
  const float* be2= (const float*)d_in[10];
  const float* m2 = (const float*)d_in[11];
  const float* v2 = (const float*)d_in[12];
  float* out = (float*)d_out;

  char* ws = (char*)d_ws;  // requires ws_size >= 98,697,216 bytes
  (void)ws_size;
  __bf16* A1  = (__bf16*)(ws);
  __bf16* H   = (__bf16*)(ws + 19267584);
  __bf16* W1b = (__bf16*)(ws + 96337920);
  __bf16* W2b = (__bf16*)(ws + 97517568);

  // pack inputs to bf16
  snn_pack_x<<<dim3(T_ * B_), dim3(224), 0, stream>>>(x, A1);
  snn_pack_w<<<dim3((CH_ * C_ / 4 + 255) / 256), dim3(256), 0, stream>>>(w1, W1b, CH_ * C_ / 4);
  snn_pack_w<<<dim3((C_ * CH_ / 4 + 255) / 256), dim3(256), 0, stream>>>(w2, W2b, C_ * CH_ / 4);

  // layer 1: GEMM + BN + LIF -> spikes
  snn_gemm1_lif<<<dim3(B_ * HWT, CH_ / 256), dim3(256), 0, stream>>>(
      A1, W1b, b1, g1, be1, m1, v1, H);

  // layer 2: GEMM + BN + LIF -> output
  snn_gemm2_lif<<<dim3(B_ * HWT, C_ / 192), dim3(192), 0, stream>>>(
      H, W2b, b2, g2, be2, m2, v2, out);
}
